// MoELayer_26749056319685
// MI455X (gfx1250) — matrix operand order repaired and checked by reading, not compiled, not run on hardware
//
#include <hip/hip_runtime.h>
#include <hip/hip_bf16.h>

typedef __attribute__((ext_vector_type(16))) _Float16 v16h;
typedef __attribute__((ext_vector_type(8)))  _Float16 v8h;
typedef __attribute__((ext_vector_type(8)))  float    v8f;
typedef __attribute__((ext_vector_type(4)))  int      v4i;

#define T_TOK   8192
#define D_MODEL 1024
#define N_EXP   8
#define CAPN    1280
#define S_SUP   4
#define DFF_N   4096

// CDNA5 async global->LDS staging if the toolchain exposes it.
#if defined(__has_builtin)
#if __has_builtin(__builtin_amdgcn_global_load_async_to_lds_b128) && \
    __has_builtin(__builtin_amdgcn_s_wait_asynccnt)
#define MOE_ASYNC_LDS 1
#endif
#endif
#ifndef MOE_ASYNC_LDS
#define MOE_ASYNC_LDS 0
#endif

#define GAS __attribute__((address_space(1)))
#define LAS __attribute__((address_space(3)))

__device__ __forceinline__ float gelu_tanh(float x) {
    float x3 = x * x * x;
    return 0.5f * x * (1.0f + tanhf(0.7978845608028654f * (x + 0.044715f * x3)));
}

// ---------------------------------------------------------------------------
// f16 WMMA GEMM: C[M,N] = A[M,K] * B[K,N] with B supplied TRANSPOSED (BT[N][K]).
// Block = 256 threads = 8 waves; block tile 128x128; wave tile 32x64
// (2 A-frags x 4 B-frags = 8 v_wmma per 32-K step). Double-buffered LDS,
// staged via GLOBAL_LOAD_ASYNC_TO_LDS_B128 when available (ASYNCcnt),
// else global_load_b128 + ds_store_b128.
// EPI 0: f32 store to Cf.   EPI 1: tanh-GELU -> f16 store to Ch.
// ---------------------------------------------------------------------------
template<int EPI>
__global__ __launch_bounds__(256)
void moe_gemm_wmma(const _Float16* __restrict__ A, const _Float16* __restrict__ BT,
                   float* __restrict__ Cf, _Float16* __restrict__ Ch,
                   int M, int N, int K, int lda, int ldbT, int ldc,
                   long sA, long sB, long sC)
{
    __shared__ __align__(64) _Float16 lA[2][128][32];   // [m][k]
    __shared__ __align__(64) _Float16 lB[2][128][32];   // [n][k]

    const int tid  = threadIdx.x;
    const int lane = tid & 31;
    const int wave = tid >> 5;
    const int wm   = wave & 3;    // m sub-block 0..3  (32 rows each)
    const int wn   = wave >> 2;   // n half 0..1       (64 cols each)
    const int bz   = blockIdx.z;

    A  += (long)bz * sA;
    BT += (long)bz * sB;
    if constexpr (EPI == 0) Cf += (long)bz * sC; else Ch += (long)bz * sC;

    const int gm0 = blockIdx.y * 128;
    const int gn0 = blockIdx.x * 128;

    // cooperative loaders: each thread moves 16 halves of A and 16 of BT per step
    const int lrow = tid >> 1;            // 0..127
    const int lcol = (tid & 1) * 16;      // 0 or 16
    const _Float16* Abase = A  + (long)(gm0 + lrow) * lda  + lcol;
    const _Float16* Bbase = BT + (long)(gn0 + lrow) * ldbT + lcol;

    const int hf   = lane >> 4;           // wave half (0/1)
    const int mrow = lane & 15;

    v8f acc[2][4] = {};

    auto load_tiles = [&](int kc, int buf) {
#if MOE_ASYNC_LDS
        __builtin_amdgcn_global_load_async_to_lds_b128(
            (GAS v4i*)(Abase + kc),     (LAS v4i*)&lA[buf][lrow][lcol],     0, 0);
        __builtin_amdgcn_global_load_async_to_lds_b128(
            (GAS v4i*)(Abase + kc + 8), (LAS v4i*)&lA[buf][lrow][lcol + 8], 0, 0);
        __builtin_amdgcn_global_load_async_to_lds_b128(
            (GAS v4i*)(Bbase + kc),     (LAS v4i*)&lB[buf][lrow][lcol],     0, 0);
        __builtin_amdgcn_global_load_async_to_lds_b128(
            (GAS v4i*)(Bbase + kc + 8), (LAS v4i*)&lB[buf][lrow][lcol + 8], 0, 0);
#else
        v8h a0 = *(const v8h*)(Abase + kc);
        v8h a1 = *(const v8h*)(Abase + kc + 8);
        v8h b0 = *(const v8h*)(Bbase + kc);
        v8h b1 = *(const v8h*)(Bbase + kc + 8);
        *(v8h*)&lA[buf][lrow][lcol]     = a0;
        *(v8h*)&lA[buf][lrow][lcol + 8] = a1;
        *(v8h*)&lB[buf][lrow][lcol]     = b0;
        *(v8h*)&lB[buf][lrow][lcol + 8] = b1;
#endif
    };
    auto wait_staging = [&]() {
#if MOE_ASYNC_LDS
        __builtin_amdgcn_s_wait_asynccnt(0);
#endif
    };

    load_tiles(0, 0);
    wait_staging();
    __syncthreads();

    const int nk = K >> 5;
    for (int step = 0; step < nk; ++step) {
        const int buf = step & 1;
        if (step + 1 < nk) load_tiles((step + 1) << 5, buf ^ 1);
        if (step + 2 < nk) {
            __builtin_prefetch(Abase + ((step + 2) << 5), 0, 1);
            __builtin_prefetch(Bbase + ((step + 2) << 5), 0, 1);
        }

        // A fragments: lane<16 holds K 0-7 & 16-23, lane>=16 holds K 8-15 & 24-31
        v16h af[2];
        #pragma unroll
        for (int mf = 0; mf < 2; ++mf) {
            v8h alo = *(const v8h*)&lA[buf][wm * 32 + mf * 16 + mrow][hf * 8];
            v8h ahi = *(const v8h*)&lA[buf][wm * 32 + mf * 16 + mrow][16 + hf * 8];
            #pragma unroll
            for (int i = 0; i < 8; ++i) { af[mf][i] = alo[i]; af[mf][i + 8] = ahi[i]; }
        }
        // B fragments: lane<16 holds K 0-7 & 16-23, lane>=16 holds K 8-15 & 24-31
        v16h bf[4];
        #pragma unroll
        for (int nf = 0; nf < 4; ++nf) {
            v8h blo = *(const v8h*)&lB[buf][wn * 64 + nf * 16 + mrow][hf * 8];
            v8h bhi = *(const v8h*)&lB[buf][wn * 64 + nf * 16 + mrow][16 + hf * 8];
            #pragma unroll
            for (int i = 0; i < 8; ++i) { bf[nf][i] = blo[i]; bf[nf][i + 8] = bhi[i]; }
        }

        #pragma unroll
        for (int mf = 0; mf < 2; ++mf)
            #pragma unroll
            for (int nf = 0; nf < 4; ++nf)
                acc[mf][nf] = __builtin_amdgcn_wmma_f32_16x16x32_f16(
                    false, af[mf], false, bf[nf], (short)0, acc[mf][nf], false, false);

        wait_staging();     // retire async ops issued this iteration (overlapped)
        __syncthreads();
    }

    // C/D layout: VGPR j -> M = j + hf*8, N = lane&15
    #pragma unroll
    for (int mf = 0; mf < 2; ++mf) {
        const int rbase = gm0 + wm * 32 + mf * 16 + hf * 8;
        #pragma unroll
        for (int j = 0; j < 8; ++j) {
            const long r = (long)(rbase + j) * ldc;
            #pragma unroll
            for (int nf = 0; nf < 4; ++nf) {
                const int ccol = gn0 + wn * 64 + nf * 16 + mrow;
                if constexpr (EPI == 0) Cf[r + ccol] = acc[mf][nf][j];
                else                    Ch[r + ccol] = (_Float16)gelu_tanh(acc[mf][nf][j]);
            }
        }
    }
}

// ---------------------------------------------------------------------------
// Router: probs = softmax(x2 @ router_w) ; one block per token
// ---------------------------------------------------------------------------
__global__ __launch_bounds__(256)
void router_kernel(const float* __restrict__ x2, const float* __restrict__ rw,
                   float* __restrict__ probs, float* __restrict__ probsT)
{
    const int t = blockIdx.x;
    const int tid = threadIdx.x, lane = tid & 31, wave = tid >> 5;
    float acc[N_EXP] = {};
    const float* xr = x2 + (long)t * D_MODEL;
    for (int d = tid; d < D_MODEL; d += 256) {
        float xv = xr[d];
        const float* wrow = rw + d * N_EXP;
        #pragma unroll
        for (int e = 0; e < N_EXP; ++e) acc[e] += xv * wrow[e];
    }
    #pragma unroll
    for (int off = 16; off; off >>= 1) {
        #pragma unroll
        for (int e = 0; e < N_EXP; ++e) acc[e] += __shfl_xor(acc[e], off, 32);
    }
    __shared__ float sred[8][N_EXP];
    if (lane == 0) { for (int e = 0; e < N_EXP; ++e) sred[wave][e] = acc[e]; }
    __syncthreads();
    if (tid < N_EXP) {
        float v = 0.f;
        for (int w = 0; w < 8; ++w) v += sred[w][tid];
        sred[0][tid] = v;
    }
    __syncthreads();
    if (tid == 0) {
        float mx = -1e30f;
        for (int e = 0; e < N_EXP; ++e) mx = fmaxf(mx, sred[0][e]);
        float pv[N_EXP]; float s = 0.f;
        for (int e = 0; e < N_EXP; ++e) { pv[e] = __expf(sred[0][e] - mx); s += pv[e]; }
        float inv = 1.0f / s;
        for (int e = 0; e < N_EXP; ++e) {
            float p = pv[e] * inv;
            probs[t * N_EXP + e] = p;
            probsT[(long)e * T_TOK + t] = p;
        }
    }
}

// ---------------------------------------------------------------------------
// Expert-choice top-CAP per expert via 32-bit radix threshold search.
// ---------------------------------------------------------------------------
__global__ __launch_bounds__(1024)
void topk_kernel(const float* __restrict__ probsT, int* __restrict__ idx,
                 float* __restrict__ gates)
{
    const int e = blockIdx.x;
    const int tid = threadIdx.x;
    __shared__ int scnt;
    __shared__ int scur;
    const float* p = probsT + (long)e * T_TOK;

    unsigned mykey[8];
    #pragma unroll
    for (int i = 0; i < 8; ++i) {
        unsigned u = __float_as_uint(p[tid + i * 1024]);
        mykey[i] = (u & 0x80000000u) ? ~u : (u | 0x80000000u);  // order-preserving
    }
    unsigned thr = 0;
    for (int bit = 31; bit >= 0; --bit) {
        const unsigned cand = thr | (1u << bit);
        if (tid == 0) scnt = 0;
        __syncthreads();
        int c = 0;
        #pragma unroll
        for (int i = 0; i < 8; ++i) c += (mykey[i] >= cand) ? 1 : 0;
        for (int off = 16; off; off >>= 1) c += __shfl_xor(c, off, 32);
        if ((tid & 31) == 0) atomicAdd(&scnt, c);
        __syncthreads();
        if (scnt >= CAPN) thr = cand;
        __syncthreads();
    }
    if (tid == 0) scur = 0;
    __syncthreads();
    #pragma unroll
    for (int i = 0; i < 8; ++i) {                       // strictly greater first
        const int t = tid + i * 1024;
        if (mykey[i] > thr) {
            int pos = atomicAdd(&scur, 1);
            idx[e * CAPN + pos]   = t;
            gates[e * CAPN + pos] = p[t];
        }
    }
    __syncthreads();
    #pragma unroll
    for (int i = 0; i < 8; ++i) {                       // fill with ties
        const int t = tid + i * 1024;
        if (mykey[i] == thr) {
            int pos = atomicAdd(&scur, 1);
            if (pos < CAPN) {
                idx[e * CAPN + pos]   = t;
                gates[e * CAPN + pos] = p[t];
            }
        }
    }
}

// ---------------------------------------------------------------------------
// TT weight reconstruction, emitted TRANSPOSED: WT[s][n][k] = sum_r c1*c2.
// ffn1: k = a*32+b (a:M1,b:M2), n = p*64+q (p:N1,q:N2)
// ---------------------------------------------------------------------------
__global__ void w1_recon(const float* __restrict__ c1, const float* __restrict__ c2,
                         _Float16* __restrict__ WT, int e)
{
    const long i = (long)blockIdx.x * blockDim.x + threadIdx.x;   // [s][4096][1024]
    if (i >= (long)S_SUP * 4096 * 1024) return;
    const int k = (int)(i & 1023),  a = k >> 5,  b = k & 31;
    const int n = (int)((i >> 10) & 4095), pp = n >> 6, q = n & 63;
    const int s = (int)(i >> 22);
    const float* Ar = c1 + ((((long)e * 4 + s) * 32 + a) * 64 + pp) * 16;   // [E,S,32,64,16]
    const float* Br = c2 + ((long)e * 4 + s) * 32768 + (long)b * 64 + q;    // [E,S,16,32,64]
    float acc = 0.f;
    #pragma unroll
    for (int r = 0; r < 16; ++r) acc += Ar[r] * Br[(long)r * 2048];
    WT[i] = (_Float16)acc;
}

// ffn2: k = a*64+b (a:N1,b:N2), n = p*32+q (p:M1,q:M2)
__global__ void w2_recon(const float* __restrict__ c1, const float* __restrict__ c2,
                         _Float16* __restrict__ WT, int e)
{
    const long i = (long)blockIdx.x * blockDim.x + threadIdx.x;   // [s][1024][4096]
    if (i >= (long)S_SUP * 1024 * 4096) return;
    const int k = (int)(i & 4095),  a = k >> 6,  b = k & 63;
    const int n = (int)((i >> 12) & 1023), pp = n >> 5, q = n & 31;
    const int s = (int)(i >> 22);
    const float* Ar = c1 + ((((long)e * 4 + s) * 64 + a) * 32 + pp) * 16;   // [E,S,64,32,16]
    const float* Br = c2 + ((long)e * 4 + s) * 32768 + (long)b * 32 + q;    // [E,S,16,64,32]
    float acc = 0.f;
    #pragma unroll
    for (int r = 0; r < 16; ++r) acc += Ar[r] * Br[(long)r * 2048];
    WT[i] = (_Float16)acc;
}

// ---------------------------------------------------------------------------
// Tiled transpose + f32->f16 cast: src[R][C] f32 -> dst[C][R] f16
// ---------------------------------------------------------------------------
__global__ __launch_bounds__(256)
void transpose_cast_kernel(const float* __restrict__ src, _Float16* __restrict__ dst,
                           int R, int C)
{
    __shared__ float tile[32][33];
    const int bx = blockIdx.x * 32;   // col base in src
    const int by = blockIdx.y * 32;   // row base in src
    const int tx = threadIdx.x & 31;
    const int ty = threadIdx.x >> 5;  // 0..7
    #pragma unroll
    for (int j = 0; j < 32; j += 8)
        tile[ty + j][tx] = src[(long)(by + ty + j) * C + bx + tx];
    __syncthreads();
    #pragma unroll
    for (int j = 0; j < 32; j += 8)
        dst[(long)(bx + ty + j) * R + by + tx] = (_Float16)tile[tx][ty + j];
}

// ---------------------------------------------------------------------------
__global__ void cast_kernel(const float* __restrict__ src, _Float16* __restrict__ dst, long n)
{
    const long i = (long)blockIdx.x * blockDim.x + threadIdx.x;
    if (i < n) dst[i] = (_Float16)src[i];
}

__global__ void zero_kernel(float* __restrict__ p, long n)
{
    const long i = (long)blockIdx.x * blockDim.x + threadIdx.x;
    if (i < n) p[i] = 0.f;
}

__global__ void gather_kernel(const float* __restrict__ x2, const int* __restrict__ idx,
                              _Float16* __restrict__ tok16, int e)
{
    const long i = (long)blockIdx.x * blockDim.x + threadIdx.x;
    if (i >= (long)CAPN * D_MODEL) return;
    const int c = (int)(i >> 10), d = (int)(i & 1023);
    const int t = idx[e * CAPN + c];
    tok16[i] = (_Float16)x2[(long)t * D_MODEL + d];
}

// ---------------------------------------------------------------------------
// Contextual gating collapse: softmax over S=4 per (token, head).
// ---------------------------------------------------------------------------
__global__ __launch_bounds__(128)
void attn_kernel(const float* __restrict__ Q, const float* __restrict__ Kk,
                 const float* __restrict__ V, const float* __restrict__ bq,
                 const float* __restrict__ bk, const float* __restrict__ bv,
                 _Float16* __restrict__ O16, int e)
{
    const int c = blockIdx.x;
    const int h = threadIdx.x >> 5;
    const int lane = threadIdx.x & 31;
    const int dbase = h * 256 + lane * 8;
    const float* bqr = bq + e * D_MODEL + dbase;
    const float* bkr = bk + e * D_MODEL + dbase;
    const float* bvr = bv + e * D_MODEL + dbase;

    float qv[8];
    #pragma unroll
    for (int u = 0; u < 8; ++u) qv[u] = Q[(long)c * D_MODEL + dbase + u] + bqr[u];

    float sc[S_SUP];
    #pragma unroll
    for (int s = 0; s < S_SUP; ++s) {
        const float* kr = Kk + ((long)c * S_SUP + s) * D_MODEL + dbase;
        float d = 0.f;
        #pragma unroll
        for (int u = 0; u < 8; ++u) d += qv[u] * (kr[u] + bkr[u]);
        for (int off = 16; off; off >>= 1) d += __shfl_xor(d, off, 32);
        sc[s] = d * 0.0625f;   // 1/sqrt(256)
    }
    float mx = fmaxf(fmaxf(sc[0], sc[1]), fmaxf(sc[2], sc[3]));
    float ssum = 0.f;
    #pragma unroll
    for (int s = 0; s < S_SUP; ++s) { sc[s] = __expf(sc[s] - mx); ssum += sc[s]; }
    const float inv = 1.f / ssum;

    float ov[8] = {};
    #pragma unroll
    for (int s = 0; s < S_SUP; ++s) {
        const float* vr = V + ((long)c * S_SUP + s) * D_MODEL + dbase;
        const float w = sc[s] * inv;
        #pragma unroll
        for (int u = 0; u < 8; ++u) ov[u] += w * (vr[u] + bvr[u]);
    }
    #pragma unroll
    for (int u = 0; u < 8; ++u) O16[(long)c * D_MODEL + dbase + u] = (_Float16)ov[u];
}

__global__ void scatter_kernel(const float* __restrict__ Out, const float* __restrict__ bo,
                               const int* __restrict__ idx, const float* __restrict__ gates,
                               float* __restrict__ out2, int e)
{
    const long i = (long)blockIdx.x * blockDim.x + threadIdx.x;
    if (i >= (long)CAPN * D_MODEL) return;
    const int c = (int)(i >> 10), d = (int)(i & 1023);
    const float g = gates[e * CAPN + c];
    const int t = idx[e * CAPN + c];
    atomicAdd(&out2[(long)t * D_MODEL + d], g * (Out[i] + bo[e * D_MODEL + d]));
}

// ---------------------------------------------------------------------------
extern "C" void kernel_launch(void* const* d_in, const int* in_sizes, int n_in,
                              void* d_out, int out_size, void* d_ws, size_t ws_size,
                              hipStream_t stream)
{
    (void)in_sizes; (void)n_in; (void)out_size;
    const float* x        = (const float*)d_in[0];
    const float* router_w = (const float*)d_in[1];
    const float* ffn1_c1  = (const float*)d_in[2];
    const float* ffn1_c2  = (const float*)d_in[3];
    const float* ffn2_c1  = (const float*)d_in[4];
    const float* ffn2_c2  = (const float*)d_in[5];
    const float* wq = (const float*)d_in[6];  const float* bq = (const float*)d_in[7];
    const float* wk = (const float*)d_in[8];  const float* bk = (const float*)d_in[9];
    const float* wv = (const float*)d_in[10]; const float* bv = (const float*)d_in[11];
    const float* wo = (const float*)d_in[12]; const float* bo = (const float*)d_in[13];
    float* out = (float*)d_out;

    char* wsb = (char*)d_ws;
    size_t off = 0;
    auto alloc = [&](size_t bytes) -> void* {
        void* p = wsb + off;
        off = (off + bytes + 255) & ~((size_t)255);
        return p;
    };
    float*     probs  = (float*)alloc((size_t)T_TOK * N_EXP * 4);
    float*     probsT = (float*)alloc((size_t)T_TOK * N_EXP * 4);
    int*       idx    = (int*)  alloc((size_t)N_EXP * CAPN * 4);
    float*     gates  = (float*)alloc((size_t)N_EXP * CAPN * 4);
    _Float16*  W1T    = (_Float16*)alloc((size_t)S_SUP * 4096 * 1024 * 2);  // [s][n][k]
    _Float16*  W2T    = (_Float16*)alloc((size_t)S_SUP * 1024 * 4096 * 2);  // [s][n][k]
    _Float16*  wq16   = (_Float16*)alloc((size_t)1024 * 1024 * 2);          // transposed
    _Float16*  wk16   = (_Float16*)alloc((size_t)1024 * 1024 * 2);
    _Float16*  wv16   = (_Float16*)alloc((size_t)1024 * 1024 * 2);
    _Float16*  wo16   = (_Float16*)alloc((size_t)1024 * 1024 * 2);
    _Float16*  tok16  = (_Float16*)alloc((size_t)CAPN * 1024 * 2);
    _Float16*  Hg16   = (_Float16*)alloc((size_t)S_SUP * CAPN * 4096 * 2);
    float*     ysup   = (float*)alloc((size_t)CAPN * 4096 * 4);
    _Float16*  ysup16 = (_Float16*)alloc((size_t)CAPN * 4096 * 2);
    float*     Qb     = (float*)alloc((size_t)CAPN * 1024 * 4);
    float*     Kb     = (float*)alloc((size_t)CAPN * S_SUP * 1024 * 4);
    float*     Vb     = (float*)alloc((size_t)CAPN * S_SUP * 1024 * 4);
    _Float16*  O16b   = (_Float16*)alloc((size_t)CAPN * 1024 * 2);
    float*     Outb   = (float*)alloc((size_t)CAPN * 1024 * 4);
    if (ws_size < off) return;   // scratch too small: refuse rather than corrupt

    zero_kernel<<<(T_TOK * D_MODEL + 255) / 256, 256, 0, stream>>>(out, (long)T_TOK * D_MODEL);
    router_kernel<<<T_TOK, 256, 0, stream>>>(x, router_w, probs, probsT);
    topk_kernel<<<N_EXP, 1024, 0, stream>>>(probsT, idx, gates);

    for (int e = 0; e < N_EXP; ++e) {
        w1_recon<<<65536, 256, 0, stream>>>(ffn1_c1, ffn1_c2, W1T, e);
        w2_recon<<<65536, 256, 0, stream>>>(ffn2_c1, ffn2_c2, W2T, e);
        transpose_cast_kernel<<<dim3(32, 32), 256, 0, stream>>>(wq + (size_t)e * 1048576, wq16, 1024, 1024);
        transpose_cast_kernel<<<dim3(32, 32), 256, 0, stream>>>(wk + (size_t)e * 1048576, wk16, 1024, 1024);
        transpose_cast_kernel<<<dim3(32, 32), 256, 0, stream>>>(wv + (size_t)e * 1048576, wv16, 1024, 1024);
        transpose_cast_kernel<<<dim3(32, 32), 256, 0, stream>>>(wo + (size_t)e * 1048576, wo16, 1024, 1024);
        gather_kernel<<<5120, 256, 0, stream>>>(x, idx, tok16, e);

        // FFN1 (+GELU, f16 out), batched over s: [1280x1024]@[1024x4096]
        moe_gemm_wmma<1><<<dim3(32, 10, 4), 256, 0, stream>>>(
            tok16, W1T, nullptr, Hg16, CAPN, 4096, 1024, 1024, 1024, 4096,
            0L, 4194304L, 5242880L);
        // FFN2, batched over s, writes ysup[c][s][d]: [1280x4096]@[4096x1024]
        moe_gemm_wmma<0><<<dim3(8, 10, 4), 256, 0, stream>>>(
            Hg16, W2T, ysup, nullptr, CAPN, 1024, 4096, 4096, 4096, 4096,
            5242880L, 4194304L, 1024L);
        cast_kernel<<<20480, 256, 0, stream>>>(ysup, ysup16, (long)CAPN * 4096);

        // Q = tok @ wq ; K = ysup @ wk ; V = ysup @ wv
        moe_gemm_wmma<0><<<dim3(8, 10, 1), 256, 0, stream>>>(
            tok16, wq16, Qb, nullptr, CAPN, 1024, 1024, 1024, 1024, 1024, 0L, 0L, 0L);
        moe_gemm_wmma<0><<<dim3(8, 40, 1), 256, 0, stream>>>(
            ysup16, wk16, Kb, nullptr, CAPN * S_SUP, 1024, 1024, 1024, 1024, 1024, 0L, 0L, 0L);
        moe_gemm_wmma<0><<<dim3(8, 40, 1), 256, 0, stream>>>(
            ysup16, wv16, Vb, nullptr, CAPN * S_SUP, 1024, 1024, 1024, 1024, 1024, 0L, 0L, 0L);

        attn_kernel<<<CAPN, 128, 0, stream>>>(Qb, Kb, Vb, bq, bk, bv, O16b, e);

        // Out = O @ wo, then gated scatter-add (+bo)
        moe_gemm_wmma<0><<<dim3(8, 10, 1), 256, 0, stream>>>(
            O16b, wo16, Outb, nullptr, CAPN, 1024, 1024, 1024, 1024, 1024, 0L, 0L, 0L);
        scatter_kernel<<<5120, 256, 0, stream>>>(Outb, bo, idx, gates, out, e);
    }
}
